// NeuronSparkMOE_83906481095358
// MI455X (gfx1250) — matrix operand order repaired and checked by reading, not compiled, not run on hardware
//
#include <hip/hip_runtime.h>
#include <hip/hip_bf16.h>

// ---------------------------------------------------------------------------
// MoE (group-limited top-4) for MI455X / gfx1250.
//  - bf16 WMMA (v_wmma_f32_16x16x32_bf16) with fp32 accumulation everywhere.
//  - One-time weight transpose+cvt to bf16 Wt[n][k]: every B fragment is two
//    contiguous b128 loads; no cvt / strided-scalar loads in inner loops.
//  - Shared-expert GEMMs stage A and B via global_load_async_to_lds_b128.
//  - Routed experts: k-outer persistent-accumulator stage-1 (A loaded once
//    per k-step, 16 WMMAs per fragment), 4-tile stage-2 passes; rolled
//    k-loops; __launch_bounds__(256,1) lifts the VGPR cap so the 128
//    accumulator VGPRs fit without scratch spills.
//  - SiLU/sigmoid use v_rcp_f32; scatter uses global_atomic_add_f32.
// ---------------------------------------------------------------------------

typedef __bf16 bf16_t;
typedef __attribute__((ext_vector_type(16))) __bf16 v16bf;
typedef __attribute__((ext_vector_type(8)))  __bf16 v8bf;
typedef __attribute__((ext_vector_type(4)))  __bf16 v4bf;
typedef __attribute__((ext_vector_type(8)))  float  v8f;

#define T_TOK 2048
#define HDIM  1024
#define NEXP  16
#define IDIM  512
#define ISH   2048

// Async 16B global->LDS copy. Generic LDS pointer's low 32 bits are the LDS
// byte offset.
__device__ __forceinline__ void copy16_g2lds(bf16_t* lds, const bf16_t* g) {
  unsigned ldsOff = (unsigned)(unsigned long long)lds;
  asm volatile("global_load_async_to_lds_b128 %0, %1, off"
               :
               : "v"(ldsOff), "v"(g)
               : "memory");
}

__device__ __forceinline__ void wait_async_lds() {
#if defined(__has_builtin) && __has_builtin(__builtin_amdgcn_s_wait_asynccnt)
  __builtin_amdgcn_s_wait_asynccnt(0);
#else
  asm volatile("s_wait_asynccnt 0x0" ::: "memory");
#endif
}

__device__ __forceinline__ v8f wmma_bf16(v16bf a, v16bf b, v8f c) {
  return __builtin_amdgcn_wmma_f32_16x16x32_bf16(false, a, false, b, (short)0, c,
                                                 false, false);
}

__device__ __forceinline__ float fast_sigmoid(float x) {
  return __builtin_amdgcn_rcpf(1.0f + __expf(-x));
}
__device__ __forceinline__ float siluf(float x) {
  return x * __builtin_amdgcn_rcpf(1.0f + __expf(-x));
}

// A-matrix 16x32 bf16 fragment (ISA 7.12.2): lane holds row (lane&15);
// element j -> col = kbase + hi*8 + (j&7) + (j>=8 ? 16 : 0): two 16B chunks.
__device__ __forceinline__ v16bf frag_a(const bf16_t* rowPtr, int kbase, int hi) {
  v8bf lo = *reinterpret_cast<const v8bf*>(rowPtr + kbase + hi * 8);
  v8bf hh = *reinterpret_cast<const v8bf*>(rowPtr + kbase + hi * 8 + 16);
  v16bf a;
#pragma unroll
  for (int i = 0; i < 8; ++i) { a[i] = lo[i]; a[i + 8] = hh[i]; }
  return a;
}

// B-matrix 32x16 fragment from transposed storage Bt[n][k] (LDS or global):
// lane holds col n; element j -> k = kbase + hi*8 + (j&7) + (j>=8 ? 16 : 0).
__device__ __forceinline__ v16bf frag_bT(const bf16_t* colRowPtr, int kbase, int hi) {
  v8bf lo = *reinterpret_cast<const v8bf*>(colRowPtr + kbase + hi * 8);
  v8bf hh = *reinterpret_cast<const v8bf*>(colRowPtr + kbase + hi * 8 + 16);
  v16bf b;
#pragma unroll
  for (int i = 0; i < 8; ++i) { b[i] = lo[i]; b[i + 8] = hh[i]; }
  return b;
}

// ---------------------------------------------------------------------------
__global__ void k_cvt_x(const float* __restrict__ x, bf16_t* __restrict__ xb) {
  int i = (blockIdx.x * 256 + threadIdx.x) * 4;
  float4 v = *reinterpret_cast<const float4*>(x + i);
  v4bf o = {(bf16_t)v.x, (bf16_t)v.y, (bf16_t)v.z, (bf16_t)v.w};
  *reinterpret_cast<v4bf*>(xb + i) = o;
}

// Batched transpose + f32->bf16: src [B][R][C] -> dst [B][C][R].
// 32x32 tiles through LDS (pad 34 -> odd dword stride, conflict-free).
__global__ void k_twcvt(const float* __restrict__ src, bf16_t* __restrict__ dst,
                        int R, int C) {
  __shared__ bf16_t t[32][34];
  size_t boff = (size_t)blockIdx.z * R * C;
  const float* s = src + boff;
  bf16_t* d = dst + boff;
  int col0 = blockIdx.x * 32, row0 = blockIdx.y * 32;
  int tx = threadIdx.x & 31, ty = threadIdx.x >> 5;   // 32 x 8
#pragma unroll
  for (int i = 0; i < 4; ++i) {
    int r = ty + i * 8;
    t[r][tx] = (bf16_t)s[(size_t)(row0 + r) * C + col0 + tx];
  }
  __syncthreads();
#pragma unroll
  for (int i = 0; i < 4; ++i) {
    int c = ty + i * 8;
    d[(size_t)(col0 + c) * R + row0 + tx] = t[tx][c];
  }
}

__global__ void k_zero_cnt(int* cnt) {
  if (threadIdx.x < NEXP) cnt[threadIdx.x] = 0;
}

// Router: 16 tokens per block, 16 threads per token (one per expert).
__global__ void k_router(const float* __restrict__ x, const float* __restrict__ gate,
                         int* __restrict__ cnt, int* __restrict__ eidx,
                         float* __restrict__ ew) {
  __shared__ float sc[16][16];
  int t = threadIdx.x;
  int lt = t >> 4, e = t & 15;
  int tok = blockIdx.x * 16 + lt;
  const float4* xr = (const float4*)(x + (size_t)tok * HDIM);
  const float4* gr = (const float4*)(gate + (size_t)e * HDIM);
  float acc = 0.f;
  for (int h = 0; h < HDIM / 4; ++h) {
    float4 a = xr[h], b = gr[h];
    acc += a.x * b.x + a.y * b.y + a.z * b.z + a.w * b.w;
  }
  sc[lt][e] = fast_sigmoid(acc);
  __syncthreads();
  if (e == 0) {
    float s[16];
#pragma unroll
    for (int i = 0; i < 16; ++i) s[i] = sc[lt][i];
    float gsc[4];
#pragma unroll
    for (int g = 0; g < 4; ++g) {
      float m1 = -1e30f, m2 = -1e30f;
#pragma unroll
      for (int j = 0; j < 4; ++j) {
        float v = s[g * 4 + j];
        if (v > m1) { m2 = m1; m1 = v; } else if (v > m2) { m2 = v; }
      }
      gsc[g] = m1 + m2;
    }
    int g1 = 0;
    for (int g = 1; g < 4; ++g) if (gsc[g] > gsc[g1]) g1 = g;
    int g2 = -1;
    for (int g = 0; g < 4; ++g) {
      if (g == g1) continue;
      if (g2 < 0 || gsc[g] > gsc[g2]) g2 = g;
    }
    unsigned mask = (0xFu << (g1 * 4)) | (0xFu << (g2 * 4));
    int idx[4]; float wv[4]; float wsum = 0.f;
    unsigned avail = mask;
#pragma unroll
    for (int k = 0; k < 4; ++k) {
      int best = 0; float bv = -1e30f;
      for (int i = 0; i < 16; ++i) {
        if (!((avail >> i) & 1)) continue;
        if (s[i] > bv) { bv = s[i]; best = i; }
      }
      avail &= ~(1u << best);
      idx[k] = best; wv[k] = bv; wsum += bv;
    }
    float scale = 2.5f / (wsum + 1e-20f);
#pragma unroll
    for (int k = 0; k < 4; ++k) {
      int ex = idx[k];
      int pos = atomicAdd(&cnt[ex], 1);
      eidx[ex * T_TOK + pos] = tok;
      ew[ex * T_TOK + pos]   = wv[k] * scale;
    }
  }
}

// ---------------------------------------------------------------------------
// Shared-expert GEMM: C[M x N] = (silu?) A[M x KDIM](bf16) @ Bt[N x KDIM](bf16).
// Block = 256 thr (8 waves); block tile 128(M) x 64(N); wave tile 32x32.
// K staged in 64-deep chunks: both A (stride 72) and Bt (stride 68) are
// async-copied to LDS with global_load_async_to_lds_b128.
template <int KDIM, int NLD, bool UP>
__global__ void k_gemm_shared(const bf16_t* __restrict__ A, const bf16_t* __restrict__ Bt,
                              void* __restrict__ Cout) {
  __shared__ bf16_t As[128 * 72];
  __shared__ bf16_t Bs[64 * 68];
  int tid = threadIdx.x;
  int wave = tid >> 5, lane = tid & 31;
  int mOff = (wave & 3) * 32, nOff = (wave >> 2) * 32;
  int mBlock = blockIdx.y * 128, nBlock = blockIdx.x * 64;
  int hi = lane >> 4, ln = lane & 15;

  v8f c00 = {}, c01 = {}, c10 = {}, c11 = {};
  for (int kc = 0; kc < KDIM; kc += 64) {
    __syncthreads();
    // Stage A: 128 rows x 64 k: 1024 x 16B chunks, 4 per thread (async).
#pragma unroll
    for (int i = 0; i < 4; ++i) {
      int c = tid + i * 256;
      int row = c >> 3, ko = (c & 7) * 8;
      copy16_g2lds(&As[row * 72 + ko],
                   A + (size_t)(mBlock + row) * KDIM + kc + ko);
    }
    // Stage Bt: 64 rows x 64 k: 512 x 16B chunks, 2 per thread (async).
#pragma unroll
    for (int i = 0; i < 2; ++i) {
      int c = tid + i * 256;
      int nrow = c >> 3, ko = (c & 7) * 8;
      copy16_g2lds(&Bs[nrow * 68 + ko],
                   Bt + (size_t)(nBlock + nrow) * KDIM + kc + ko);
    }
    wait_async_lds();
    __syncthreads();
#pragma unroll
    for (int kk = 0; kk < 64; kk += 32) {
      v16bf a0 = frag_a(&As[(mOff + ln) * 72], kk, hi);
      v16bf a1 = frag_a(&As[(mOff + 16 + ln) * 72], kk, hi);
      v16bf b0 = frag_bT(&Bs[(nOff + ln) * 68], kk, hi);
      v16bf b1 = frag_bT(&Bs[(nOff + 16 + ln) * 68], kk, hi);
      c00 = wmma_bf16(a0, b0, c00);
      c01 = wmma_bf16(a0, b1, c01);
      c10 = wmma_bf16(a1, b0, c10);
      c11 = wmma_bf16(a1, b1, c11);
    }
  }
  // D layout: element r -> row = r + hi*8 (+16 for second band), col = ln.
#pragma unroll
  for (int r = 0; r < 8; ++r) {
    int row0 = mBlock + mOff + hi * 8 + r, row1 = row0 + 16;
    int col0 = nBlock + nOff + ln, col1 = col0 + 16;
    if (UP) {
      bf16_t* o = (bf16_t*)Cout;
      o[(size_t)row0 * NLD + col0] = (bf16_t)siluf(c00[r]);
      o[(size_t)row0 * NLD + col1] = (bf16_t)siluf(c01[r]);
      o[(size_t)row1 * NLD + col0] = (bf16_t)siluf(c10[r]);
      o[(size_t)row1 * NLD + col1] = (bf16_t)siluf(c11[r]);
    } else {
      float* o = (float*)Cout;
      o[(size_t)row0 * NLD + col0] = c00[r];
      o[(size_t)row0 * NLD + col1] = c01[r];
      o[(size_t)row1 * NLD + col0] = c10[r];
      o[(size_t)row1 * NLD + col1] = c11[r];
    }
  }
}

// ---------------------------------------------------------------------------
// Fused routed expert: per (expert, 64-token tile):
//   stage 1: U = silu(Xg @ Wu_e) -> 64KB LDS. k-outer: 16 persistent
//            accumulator tiles per wave; A fragment loaded once per k-step.
//   stage 2: out += (U @ Wd_e) * routing_weight; 4-tile passes, rolled k-loop.
// __launch_bounds__(256, 1): allow the full VGPR budget (no scratch spills).
__global__ void __launch_bounds__(256, 1)
k_moe(const bf16_t* __restrict__ xb,
      const bf16_t* __restrict__ wut,   // [16][512][1024]  (n,k)
      const bf16_t* __restrict__ wdt,   // [16][1024][512]  (n,k)
      const int* __restrict__ cnt, const int* __restrict__ eidx,
      const float* __restrict__ ew, float* __restrict__ out) {
  __shared__ bf16_t U[64 * IDIM];   // 65536 B; addr = ((ct*64 + row)*16 + cin)
  int e = blockIdx.y;
  int base = blockIdx.x * 64;
  int n = cnt[e];
  if (base >= n) return;
  const bf16_t* Wu = wut + (size_t)e * IDIM * HDIM;
  const bf16_t* Wd = wdt + (size_t)e * HDIM * IDIM;
  int wave = threadIdx.x >> 5, lane = threadIdx.x & 31;
  int mb = wave & 3, nh = wave >> 2;
  int hi = lane >> 4, ln = lane & 15;

  int slotA = base + mb * 16 + ln;
  int tokA = (slotA < n) ? eidx[e * T_TOK + slotA] : 0;
  const bf16_t* arow = xb + (size_t)tokA * HDIM;

  // ---- Stage 1: k-outer, 16 persistent accumulator tiles (128 VGPRs). ----
  {
    v8f acc[16];
#pragma unroll
    for (int t = 0; t < 16; ++t) acc[t] = (v8f){};
    const bf16_t* wbase = Wu + (size_t)(nh * 256 + ln) * HDIM;
#pragma clang loop unroll(disable)
    for (int k = 0; k < HDIM; k += 32) {
      v16bf a = frag_a(arow, k, hi);
#pragma unroll
      for (int t = 0; t < 16; ++t) {
        v16bf b = frag_bT(wbase + (size_t)t * 16 * HDIM, k, hi);
        acc[t] = wmma_bf16(a, b, acc[t]);
      }
    }
#pragma unroll
    for (int t = 0; t < 16; ++t) {
      int ct = (nh * 256 + t * 16) >> 4;
#pragma unroll
      for (int r = 0; r < 8; ++r) {
        int row = mb * 16 + hi * 8 + r;
        U[(ct * 64 + row) * 16 + ln] = (bf16_t)siluf(acc[t][r]);
      }
    }
  }
  __syncthreads();

  // Scatter metadata: this lane's 8 output rows (fixed for all passes).
  int sIdx[8]; float sW[8];
#pragma unroll
  for (int r = 0; r < 8; ++r) {
    int slot = base + mb * 16 + hi * 8 + r;
    bool v = slot < n;
    sIdx[r] = v ? eidx[e * T_TOK + slot] : 0;
    sW[r]   = v ? ew[e * T_TOK + slot] : 0.0f;
  }

  // ---- Stage 2: 4-tile passes (64 cols), rolled k-loop (no spills). ----
  int rowA = mb * 16 + ln;
#pragma clang loop unroll(disable)
  for (int pass = 0; pass < 8; ++pass) {
    int nb = nh * 512 + pass * 64;
    const bf16_t* wdbase = Wd + (size_t)(nb + ln) * IDIM;
    v8f c[4];
#pragma unroll
    for (int t = 0; t < 4; ++t) c[t] = (v8f){};
#pragma clang loop unroll(disable)
    for (int k = 0; k < IDIM; k += 32) {
      v8bf lo = *reinterpret_cast<const v8bf*>(&U[((k >> 4) * 64 + rowA) * 16 + hi * 8]);
      v8bf hh = *reinterpret_cast<const v8bf*>(&U[(((k >> 4) + 1) * 64 + rowA) * 16 + hi * 8]);
      v16bf a;
#pragma unroll
      for (int i = 0; i < 8; ++i) { a[i] = lo[i]; a[i + 8] = hh[i]; }
#pragma unroll
      for (int t = 0; t < 4; ++t) {
        v16bf b = frag_bT(wdbase + (size_t)t * 16 * IDIM, k, hi);
        c[t] = wmma_bf16(a, b, c[t]);
      }
    }
#pragma unroll
    for (int t = 0; t < 4; ++t) {
#pragma unroll
      for (int r = 0; r < 8; ++r) {
        size_t ro = (size_t)sIdx[r] * HDIM;
        unsafeAtomicAdd(&out[ro + nb + t * 16 + ln], c[t][r] * sW[r]);
      }
    }
  }
}

// ---------------------------------------------------------------------------
extern "C" void kernel_launch(void* const* d_in, const int* in_sizes, int n_in,
                              void* d_out, int out_size, void* d_ws, size_t ws_size,
                              hipStream_t stream) {
  const float* x  = (const float*)d_in[0];   // [2,1024,1024]
  const float* gw = (const float*)d_in[1];   // [16,1024]
  const float* eu = (const float*)d_in[2];   // [16,1024,512]
  const float* ed = (const float*)d_in[3];   // [16,512,1024]
  const float* su = (const float*)d_in[4];   // [1024,2048]
  const float* sd = (const float*)d_in[5];   // [2048,1024]
  float* out = (float*)d_out;

  char* ws = (char*)d_ws;
  bf16_t* xb   = (bf16_t*)(ws);                 // [2048][1024]       4 MB
  bf16_t* sact = (bf16_t*)(ws + 4194304);       // [2048][2048]       8 MB
  bf16_t* sut  = (bf16_t*)(ws + 12582912);      // [2048][1024]       4 MB
  bf16_t* sdt  = (bf16_t*)(ws + 16777216);      // [1024][2048]       4 MB
  bf16_t* wut  = (bf16_t*)(ws + 20971520);      // [16][512][1024]   16 MB
  bf16_t* wdt  = (bf16_t*)(ws + 37748736);      // [16][1024][512]   16 MB
  int*    cnt  = (int*)(ws + 54525952);         // 64 B (+pad)
  int*    eidx = (int*)(ws + 54526208);         // 128 KB
  float*  ewt  = (float*)(ws + 54657280);       // 128 KB

  k_cvt_x<<<(T_TOK * HDIM) / 1024, 256, 0, stream>>>(x, xb);
  k_zero_cnt<<<1, 32, 0, stream>>>(cnt);
  k_router<<<T_TOK / 16, 256, 0, stream>>>(x, gw, cnt, eidx, ewt);
  // One-time weight transpose+cvt to bf16 Wt[n][k].
  k_twcvt<<<dim3(ISH / 32, HDIM / 32, 1),    256, 0, stream>>>(su, sut, HDIM, ISH);
  k_twcvt<<<dim3(HDIM / 32, ISH / 32, 1),    256, 0, stream>>>(sd, sdt, ISH, HDIM);
  k_twcvt<<<dim3(IDIM / 32, HDIM / 32, NEXP), 256, 0, stream>>>(eu, wut, HDIM, IDIM);
  k_twcvt<<<dim3(HDIM / 32, IDIM / 32, NEXP), 256, 0, stream>>>(ed, wdt, IDIM, HDIM);

  k_gemm_shared<HDIM, ISH, true><<<dim3(ISH / 64, T_TOK / 128), 256, 0, stream>>>(
      xb, sut, (void*)sact);
  k_gemm_shared<ISH, HDIM, false><<<dim3(HDIM / 64, T_TOK / 128), 256, 0, stream>>>(
      sact, sdt, (void*)out);
  k_moe<<<dim3(T_TOK / 64, NEXP), 256, 0, stream>>>(xb, wut, wdt, cnt, eidx, ewt, out);
}
